// Deepv2_base_74741020885606
// MI455X (gfx1250) — matrix operand order repaired and checked by reading, not compiled, not run on hardware
//
#include <hip/hip_runtime.h>
#include <math.h>

// ---------------------------------------------------------------------------
// Types for CDNA5 WMMA (wave32): V_WMMA_F32_16X16X32_BF16
// ---------------------------------------------------------------------------
typedef __attribute__((ext_vector_type(16))) __bf16 v16bf;
typedef __attribute__((ext_vector_type(8)))  __bf16 v8bf;
typedef __attribute__((ext_vector_type(8)))  float  v8f;
typedef __attribute__((ext_vector_type(4)))  unsigned v4u;
typedef __attribute__((ext_vector_type(8)))  int      v8i;
typedef __attribute__((ext_vector_type(4)))  int      v4i;

#define B_BATCH 16
#define S_LOC   128
#define T_LOC   256
#define D_DIM   1024

// Pack two f32 into one dword holding two bf16 (element0 in low half).
__device__ __forceinline__ unsigned pack_rne(float f0, float f1) {
    unsigned u0 = __builtin_bit_cast(unsigned, f0);
    unsigned u1 = __builtin_bit_cast(unsigned, f1);
    u0 += 0x7FFFu + ((u0 >> 16) & 1u);
    u1 += 0x7FFFu + ((u1 >> 16) & 1u);
    return __builtin_amdgcn_perm(u1, u0, 0x07060302u);   // {hi16(u1), hi16(u0)}
}
// Truncating version: a single v_perm_b32 for two elements (hot B paths).
__device__ __forceinline__ unsigned pack_trunc(float f0, float f1) {
    return __builtin_amdgcn_perm(__builtin_bit_cast(unsigned, f1),
                                 __builtin_bit_cast(unsigned, f0), 0x07060302u);
}

__device__ __forceinline__ float wave_sum(float v) {
#pragma unroll
    for (int off = 16; off; off >>= 1) v += __shfl_xor(v, off, 32);
    return v;
}
__device__ __forceinline__ float wave_max(float v) {
#pragma unroll
    for (int off = 16; off; off >>= 1) v = fmaxf(v, __shfl_xor(v, off, 32));
    return v;
}

// A fragment (16x32 bf16) from bf16 LDS, row-major, pitch elements.
__device__ __forceinline__ v16bf load_a_frag(const __bf16* sA, int pitch, int m, int half, int kk) {
    union { v16bf v; v8bf h[2]; } u;
    const __bf16* p = sA + m * pitch + kk + half * 16;
    u.h[0] = *(const v8bf*)p;          // ds_load_b128
    u.h[1] = *(const v8bf*)(p + 8);    // ds_load_b128
    return u.v;
}

// A fragment from f32 LDS (TDM-staged tile), converted at read time.
__device__ __forceinline__ v16bf load_a_frag_f32lds(const float* sAf, int pitch,
                                                    int m, int half, int kk) {
    union { v16bf v; unsigned u[8]; } r;
    const float* p = sAf + m * pitch + kk + half * 16;
    float4 a0 = *(const float4*)p;         // ds_load_b128 x4
    float4 a1 = *(const float4*)(p + 4);
    float4 a2 = *(const float4*)(p + 8);
    float4 a3 = *(const float4*)(p + 12);
    r.u[0] = pack_rne(a0.x, a0.y); r.u[1] = pack_rne(a0.z, a0.w);
    r.u[2] = pack_rne(a1.x, a1.y); r.u[3] = pack_rne(a1.z, a1.w);
    r.u[4] = pack_rne(a2.x, a2.y); r.u[5] = pack_rne(a2.z, a2.w);
    r.u[6] = pack_rne(a3.x, a3.y); r.u[7] = pack_rne(a3.z, a3.w);
    return r.v;
}

// B fragment from pre-swizzled bf16 weights: 32 contiguous bytes per lane.
__device__ __forceinline__ v16bf load_b_swz(const __bf16* wp) {
    union { v16bf v; v8bf h[2]; } u;
    u.h[0] = *(const v8bf*)wp;         // global_load_b128
    u.h[1] = *(const v8bf*)(wp + 8);   // global_load_b128
    return u.v;
}

// B fragment from f32, column-strided (on-the-fly convert, perm-packed).
__device__ __forceinline__ v16bf load_b_strided(const float* wp, size_t pitch) {
    union { v16bf v; unsigned u[8]; } r;
#pragma unroll
    for (int j = 0; j < 8; ++j)
        r.u[j] = pack_trunc(wp[(size_t)(2 * j) * pitch], wp[(size_t)(2 * j + 1) * pitch]);
    return r.v;
}

// B fragment from f32, 16 contiguous values (NT attention path).
__device__ __forceinline__ v16bf load_b_contig16(const float* wp) {
    union { v16bf v; unsigned u[8]; } r;
    const float4* p4 = (const float4*)wp;   // 64B-aligned by construction
    float4 q0 = p4[0], q1 = p4[1], q2 = p4[2], q3 = p4[3];
    r.u[0] = pack_trunc(q0.x, q0.y); r.u[1] = pack_trunc(q0.z, q0.w);
    r.u[2] = pack_trunc(q1.x, q1.y); r.u[3] = pack_trunc(q1.z, q1.w);
    r.u[4] = pack_trunc(q2.x, q2.y); r.u[5] = pack_trunc(q2.z, q2.w);
    r.u[6] = pack_trunc(q3.x, q3.y); r.u[7] = pack_trunc(q3.z, q3.w);
    return r.v;
}

// Stage a 16 x KC f32 tile into LDS as bf16 (dword-packed writes) and
// prefetch the next K-chunk (global_prefetch).
template <int KC>
__device__ __forceinline__ void stage_A(const float* __restrict__ Abase, int apitch,
                                        int kc, unsigned* __restrict__ sAu, int tid) {
    constexpr int HKC = KC / 2;
#pragma unroll 2
    for (int i = tid; i < 16 * HKC; i += 256) {
        int r = i / HKC;
        int c = (i - r * HKC) * 2;
        const float* ap = Abase + (size_t)r * apitch + kc + c;
        const float2 f = *(const float2*)ap;
        __builtin_prefetch((const void*)(ap + KC), 0, 0);   // next chunk
        sAu[i] = pack_rne(f.x, f.y);
    }
}

// ---------------------------------------------------------------------------
// One-time (per launch) weight conversion f32[K,N] -> bf16 B-fragment layout:
// out[(kb*(N/16)+nb)*512 + lane*16 + j]  (one wave converts one 32x16 tile)
// ---------------------------------------------------------------------------
__global__ void __launch_bounds__(256) swizzle_w_kernel(
    const float* __restrict__ W, __bf16* __restrict__ out, int K, int N)
{
    int gw   = (blockIdx.x * 256 + threadIdx.x) >> 5;   // global wave id = tile id
    int lane = threadIdx.x & 31;
    int ntn  = N >> 4;
    int nt   = (K >> 5) * ntn;
    if (gw >= nt) return;
    int kb = gw / ntn, nb = gw - kb * ntn;
    int m = lane & 15, half = lane >> 4;
    const float* wp = W + ((size_t)kb * 32 + half * 16) * N + nb * 16 + m;
    union { v8bf h[2]; unsigned u[8]; } r;
#pragma unroll
    for (int j = 0; j < 8; ++j)
        r.u[j] = pack_rne(wp[(size_t)(2 * j) * N], wp[(size_t)(2 * j + 1) * N]);
    __bf16* op = out + (size_t)gw * 512 + lane * 16;
    *(v8bf*)op       = r.h[0];
    *(v8bf*)(op + 8) = r.h[1];
}

// ---------------------------------------------------------------------------
// flatten_pad: ragged rows (sorted by batch id) -> dense [B, local, D] gather
// ---------------------------------------------------------------------------
__device__ __forceinline__ int lower_bound_i(const int* a, int n, int v) {
    int lo = 0, hi = n;
    while (lo < hi) { int mid = (lo + hi) >> 1; if (a[mid] < v) lo = mid + 1; else hi = mid; }
    return lo;
}

__global__ void __launch_bounds__(256) flatten_kernel(
    const float* __restrict__ feats, const int* __restrict__ bidx,
    int nrows, int local, float* __restrict__ out)
{
    int blk  = blockIdx.x;
    int b    = blk / local;
    int slot = blk % local;
    int s0 = lower_bound_i(bidx, nrows, b);
    int s1 = lower_bound_i(bidx, nrows, b + 1);
    int src = s0 + slot;
    float4* dst = (float4*)(out + (size_t)blk * D_DIM);
    int t = threadIdx.x;
    if (src < s1) {
        const float4* sp = (const float4*)(feats + (size_t)src * D_DIM);
        dst[t] = sp[t];
    } else {
        dst[t] = make_float4(0.f, 0.f, 0.f, 0.f);
    }
}

// ---------------------------------------------------------------------------
// Fused  Y = relu(LN(A @ W)) with pre-swizzled bf16 W. 16-row tile / block,
// 8 waves x TN 16x16 column tiles (TN*16*8 == N), LN over full row.
// ---------------------------------------------------------------------------
template <int N, int TN, int KC>
__global__ void __launch_bounds__(256) gemm_ln_relu_kernel(
    const float* __restrict__ A, int apitch,
    const __bf16* __restrict__ Wswz,
    const float* __restrict__ gam, const float* __restrict__ bet,
    float* __restrict__ out, int out_stride, int out_coloff, int K)
{
    static_assert(TN * 16 * 8 == N, "tile cover");
    __shared__ float smem[16 * N];                      // Y stage; aliases A stage
    __bf16*   sA  = (__bf16*)smem;
    unsigned* sAu = (unsigned*)smem;

    const int tid  = threadIdx.x;
    const int wave = tid >> 5, lane = tid & 31, m = lane & 15, half = lane >> 4;
    const int row0 = blockIdx.x * 16;
    const int n0   = wave * (TN * 16);
    const int ntn  = N >> 4;

    const v8f zero8 = {0.f,0.f,0.f,0.f,0.f,0.f,0.f,0.f};
    v8f acc[TN];
#pragma unroll
    for (int t = 0; t < TN; ++t) acc[t] = zero8;

    const float* Abase = A + (size_t)row0 * apitch;

    for (int kc = 0; kc < K; kc += KC) {
        __syncthreads();
        stage_A<KC>(Abase, apitch, kc, sAu, tid);
        __syncthreads();
#pragma unroll 2
        for (int kk = 0; kk < KC; kk += 32) {
            v16bf afrag = load_a_frag(sA, KC, m, half, kk);
            int kb = (kc + kk) >> 5;
#pragma unroll
            for (int t = 0; t < TN; ++t) {
                const __bf16* wp = Wswz + ((size_t)kb * ntn + (n0 >> 4) + t) * 512 + lane * 16;
                v16bf bfrag = load_b_swz(wp);
                acc[t] = __builtin_amdgcn_wmma_f32_16x16x32_bf16(
                    false, afrag, false, bfrag, (short)0, acc[t], false, false);
            }
        }
    }

    __syncthreads();
#pragma unroll
    for (int t = 0; t < TN; ++t)
#pragma unroll
        for (int r = 0; r < 8; ++r)
            smem[(r + half * 8) * N + n0 + t * 16 + m] = acc[t][r];
    __syncthreads();

    for (int r = wave * 2; r < wave * 2 + 2; ++r) {
        float s = 0.f, s2 = 0.f;
        for (int n = lane; n < N; n += 32) { float v = smem[r * N + n]; s += v; s2 += v * v; }
        s = wave_sum(s); s2 = wave_sum(s2);
        float mean = s * (1.0f / N);
        float var  = s2 * (1.0f / N) - mean * mean;
        float rs   = rsqrtf(var + 1e-5f);
        float* orow = out + (size_t)(row0 + r) * out_stride + out_coloff;
        for (int n = lane; n < N; n += 32)
            orow[n] = fmaxf((smem[r * N + n] - mean) * rs * gam[n] + bet[n], 0.f);
    }
}

// ---------------------------------------------------------------------------
// Fused attention scores + softmax: attn = softmax(q[b] @ k[b]^T) per batch.
// ---------------------------------------------------------------------------
__global__ void __launch_bounds__(256) attn_kernel(
    const float* __restrict__ q, const float* __restrict__ kmat,
    float* __restrict__ attn)
{
    constexpr int N = T_LOC, TN = 2, KC = 256, K = D_DIM;
    __shared__ float smem[16 * N];
    __bf16*   sA  = (__bf16*)smem;
    unsigned* sAu = (unsigned*)smem;

    const int tid = threadIdx.x, wave = tid >> 5, lane = tid & 31, m = lane & 15, half = lane >> 4;
    const int b = blockIdx.y, row0 = blockIdx.x * 16;
    const float* Ab = q    + (size_t)(b * S_LOC + row0) * D_DIM;
    const float* Kb = kmat + (size_t)b * T_LOC * D_DIM;
    const int n0 = wave * (TN * 16);

    const v8f zero8 = {0.f,0.f,0.f,0.f,0.f,0.f,0.f,0.f};
    v8f acc[TN];
#pragma unroll
    for (int t = 0; t < TN; ++t) acc[t] = zero8;

    for (int kc = 0; kc < K; kc += KC) {
        __syncthreads();
        stage_A<KC>(Ab, D_DIM, kc, sAu, tid);
        __syncthreads();
#pragma unroll 2
        for (int kk = 0; kk < KC; kk += 32) {
            v16bf afrag = load_a_frag(sA, KC, m, half, kk);
#pragma unroll
            for (int t = 0; t < TN; ++t) {
                int n = n0 + t * 16 + m;                         // velo index
                const float* kp = Kb + (size_t)n * D_DIM + kc + kk + half * 16;
                v16bf bfrag = load_b_contig16(kp);
                acc[t] = __builtin_amdgcn_wmma_f32_16x16x32_bf16(
                    false, afrag, false, bfrag, (short)0, acc[t], false, false);
            }
        }
    }

    __syncthreads();
#pragma unroll
    for (int t = 0; t < TN; ++t)
#pragma unroll
        for (int r = 0; r < 8; ++r)
            smem[(r + half * 8) * N + n0 + t * 16 + m] = acc[t][r];
    __syncthreads();

    for (int r = wave * 2; r < wave * 2 + 2; ++r) {
        float mx = -3.402823466e38f;
        for (int n = lane; n < N; n += 32) mx = fmaxf(mx, smem[r * N + n]);
        mx = wave_max(mx);
        float s = 0.f;
        for (int n = lane; n < N; n += 32) {
            float e = __expf(smem[r * N + n] - mx);
            smem[r * N + n] = e; s += e;
        }
        s = wave_sum(s);
        float inv = 1.f / s;
        float* arow = attn + ((size_t)b * S_LOC + row0 + r) * N;
        for (int n = lane; n < N; n += 32) arow[n] = smem[r * N + n] * inv;
    }
}

// ---------------------------------------------------------------------------
// ctx = attn @ v1 per batch; output into x[b, s*1024 + 512 + c]
// ---------------------------------------------------------------------------
__global__ void __launch_bounds__(256) ctx_kernel(
    const float* __restrict__ attn, const float* __restrict__ v1,
    float* __restrict__ x)
{
    constexpr int N = 512, TN = 4, KC = 256;
    __shared__ __bf16 sA[16 * KC];

    const int tid = threadIdx.x, wave = tid >> 5, lane = tid & 31, m = lane & 15, half = lane >> 4;
    const int b = blockIdx.y, row0 = blockIdx.x * 16;
    const float* Ab = attn + (size_t)(b * S_LOC + row0) * T_LOC;
    const float* Wb = v1   + (size_t)b * T_LOC * N;
    const int n0 = wave * (TN * 16);

    const v8f zero8 = {0.f,0.f,0.f,0.f,0.f,0.f,0.f,0.f};
    v8f acc[TN];
#pragma unroll
    for (int t = 0; t < TN; ++t) acc[t] = zero8;

    stage_A<KC>(Ab, T_LOC, 0, (unsigned*)sA, tid);
    __syncthreads();
#pragma unroll 2
    for (int kk = 0; kk < KC; kk += 32) {
        v16bf afrag = load_a_frag(sA, KC, m, half, kk);
#pragma unroll
        for (int t = 0; t < TN; ++t) {
            int n = n0 + t * 16 + m;
            const float* wp = Wb + (size_t)(kk + half * 16) * N + n;
            v16bf bfrag = load_b_strided(wp, N);
            acc[t] = __builtin_amdgcn_wmma_f32_16x16x32_bf16(
                false, afrag, false, bfrag, (short)0, acc[t], false, false);
        }
    }

    float* xb = x + (size_t)b * (S_LOC * D_DIM);
#pragma unroll
    for (int t = 0; t < TN; ++t)
#pragma unroll
        for (int r = 0; r < 8; ++r)
            xb[(size_t)(row0 + r + half * 8) * D_DIM + 512 + n0 + t * 16 + m] = acc[t][r];
}

// ---------------------------------------------------------------------------
// Split-K GEMM for the 512MB W3a matmul (HBM-bound). A-tile staged into LDS
// by the Tensor Data Mover (TDM, 2D descriptor), synced via TENSORcnt.
// ---------------------------------------------------------------------------
template <int KSLICE, int KC>
__global__ void __launch_bounds__(256) gemm3a_splitk_kernel(
    const float* __restrict__ A,      // [16, 131072] f32
    const float* __restrict__ W,      // [131072, 1024] f32
    float* __restrict__ ypart)        // [16, 1024] f32 (pre-zeroed)
{
    constexpr int N = 1024, TN = 8;
    constexpr unsigned KTOT = 131072u;
    __shared__ float sAf[16 * KC];    // f32 tile staged by TDM

    const int tid = threadIdx.x, wave = tid >> 5, lane = tid & 31, m = lane & 15, half = lane >> 4;
    const int kbase = blockIdx.x * KSLICE;
    const int n0 = wave * (TN * 16);

    const unsigned lds_off = (unsigned)(size_t)&sAf[0];

    const v8f zero8 = {0.f,0.f,0.f,0.f,0.f,0.f,0.f,0.f};
    v8f acc[TN];
#pragma unroll
    for (int t = 0; t < TN; ++t) acc[t] = zero8;

    for (int kc = 0; kc < KSLICE; kc += KC) {
        __syncthreads();                       // prior readers done with sAf
        if (wave == 0) {
            // Tensor DMA descriptor: 2D tile [tile_dim1=16 rows x tile_dim0=KC elems],
            // 4-byte elements, row stride = KTOT, dest = sAf (packed rows).
            unsigned long long ga = (unsigned long long)(const void*)(A + (size_t)kbase + kc);
            v4u g0 = { 1u,                                    // count=1 (valid user D#)
                       lds_off,                               // lds_addr
                       (unsigned)(ga & 0xFFFFFFFFu),          // global_addr[31:0]
                       (unsigned)((ga >> 32) & 0x01FFFFFFu)   // global_addr[56:32]
                       | 0x80000000u };                       // type=2 ("image")
            v8i g1 = { (int)0x00020000u,                      // data_size=2 (4B); wg_mask=0
                       0,                                     // no atomic barrier; dim0 lo16=0
                       (int)((KTOT >> 16) | (16u << 16)),     // tensor_dim0 hi; tensor_dim1=16
                       (int)((unsigned)KC << 16),             // tile_dim0 = KC
                       16,                                    // tile_dim1=16, tile_dim2=0
                       (int)KTOT,                             // tensor_dim0_stride lo32
                       0, 0 };                                // stride hi, dim1_stride
            v4i gz4 = { 0, 0, 0, 0 };
            v8i gz8 = { 0, 0, 0, 0, 0, 0, 0, 0 };
            __builtin_amdgcn_tensor_load_to_lds(g0, g1, gz4, gz4, gz8, 0);
            __builtin_amdgcn_s_wait_tensorcnt(0);
        }
        __syncthreads();                       // tile visible to all waves
#pragma unroll 2
        for (int kk = 0; kk < KC; kk += 32) {
            v16bf afrag = load_a_frag_f32lds(sAf, KC, m, half, kk);
#pragma unroll
            for (int t = 0; t < TN; ++t) {
                int n = n0 + t * 16 + m;
                const float* wp = W + (size_t)(kbase + kc + kk + half * 16) * N + n;
                v16bf bfrag = load_b_strided(wp, N);
                acc[t] = __builtin_amdgcn_wmma_f32_16x16x32_bf16(
                    false, afrag, false, bfrag, (short)0, acc[t], false, false);
            }
        }
    }

#pragma unroll
    for (int t = 0; t < TN; ++t)
#pragma unroll
        for (int r = 0; r < 8; ++r)
            atomicAdd(&ypart[(r + half * 8) * N + n0 + t * 16 + m], acc[t][r]);
}

// ---------------------------------------------------------------------------
// LN + ReLU over 16 rows of [16, N] (post split-K epilogue)
// ---------------------------------------------------------------------------
__global__ void __launch_bounds__(256) ln_relu_rows_kernel(
    const float* __restrict__ y, const float* __restrict__ gam,
    const float* __restrict__ bet, float* __restrict__ out, int N)
{
    const int wave = threadIdx.x >> 5, lane = threadIdx.x & 31;
    for (int r = wave * 2; r < wave * 2 + 2; ++r) {
        const float* yr = y + (size_t)r * N;
        float s = 0.f, s2 = 0.f;
        for (int n = lane; n < N; n += 32) { float v = yr[n]; s += v; s2 += v * v; }
        s = wave_sum(s); s2 = wave_sum(s2);
        float mean = s / N, var = s2 / N - mean * mean;
        float rs = rsqrtf(var + 1e-5f);
        for (int n = lane; n < N; n += 32)
            out[(size_t)r * N + n] = fmaxf((yr[n] - mean) * rs * gam[n] + bet[n], 0.f);
    }
}

__global__ void zero_kernel(float* __restrict__ p, int n) {
    int i = blockIdx.x * blockDim.x + threadIdx.x;
    if (i < n) p[i] = 0.f;
}

// ---------------------------------------------------------------------------
// Heads: quat = normalize(x@Wr + br), trans = x@Wt + bt. One wave per batch.
// ---------------------------------------------------------------------------
__global__ void __launch_bounds__(32) heads_kernel(
    const float* __restrict__ x,
    const float* __restrict__ Wr, const float* __restrict__ br,
    const float* __restrict__ Wt, const float* __restrict__ bt,
    float* __restrict__ out)
{
    const int b = blockIdx.x, lane = threadIdx.x;
    const float* xr = x + (size_t)b * D_DIM;
    float d[7] = {0.f,0.f,0.f,0.f,0.f,0.f,0.f};
    for (int k = lane; k < D_DIM; k += 32) {
        float xv = xr[k];
#pragma unroll
        for (int j = 0; j < 4; ++j) d[j]     += xv * Wr[(size_t)k * 4 + j];
#pragma unroll
        for (int j = 0; j < 3; ++j) d[4 + j] += xv * Wt[(size_t)k * 3 + j];
    }
#pragma unroll
    for (int j = 0; j < 7; ++j) d[j] = wave_sum(d[j]);
    if (lane == 0) {
        float q0 = d[0] + br[0], q1 = d[1] + br[1], q2 = d[2] + br[2], q3 = d[3] + br[3];
        float nrm = fmaxf(sqrtf(q0*q0 + q1*q1 + q2*q2 + q3*q3), 1e-12f);
        out[b * 4 + 0] = q0 / nrm;
        out[b * 4 + 1] = q1 / nrm;
        out[b * 4 + 2] = q2 / nrm;
        out[b * 4 + 3] = q3 / nrm;
        out[B_BATCH * 4 + b * 3 + 0] = d[4] + bt[0];
        out[B_BATCH * 4 + b * 3 + 1] = d[5] + bt[1];
        out[B_BATCH * 4 + b * 3 + 2] = d[6] + bt[2];
    }
}

// ---------------------------------------------------------------------------
// Launcher
// ---------------------------------------------------------------------------
extern "C" void kernel_launch(void* const* d_in, const int* in_sizes, int n_in,
                              void* d_out, int out_size, void* d_ws, size_t ws_size,
                              hipStream_t stream) {
    const float* sgm_feats  = (const float*)d_in[0];
    const float* velo_feats = (const float*)d_in[1];
    const int*   sgm_bidx   = (const int*)d_in[2];
    const int*   velo_bidx  = (const int*)d_in[3];
    const float* Wq  = (const float*)d_in[4];
    const float* gq  = (const float*)d_in[5];
    const float* bq  = (const float*)d_in[6];
    const float* Wk  = (const float*)d_in[7];
    const float* gk  = (const float*)d_in[8];
    const float* bk  = (const float*)d_in[9];
    const float* Wv1 = (const float*)d_in[10];
    const float* gv1 = (const float*)d_in[11];
    const float* bv1 = (const float*)d_in[12];
    const float* Wv2 = (const float*)d_in[13];
    const float* gv2 = (const float*)d_in[14];
    const float* bv2 = (const float*)d_in[15];
    const float* W3a = (const float*)d_in[16];
    const float* g3a = (const float*)d_in[17];
    const float* b3a = (const float*)d_in[18];
    const float* W3b = (const float*)d_in[19];
    const float* g3b = (const float*)d_in[20];
    const float* b3b = (const float*)d_in[21];
    const float* Wr  = (const float*)d_in[22];
    const float* br  = (const float*)d_in[23];
    const float* Wt  = (const float*)d_in[24];
    const float* bt  = (const float*)d_in[25];

    const int n_sgm  = in_sizes[0] / D_DIM;
    const int n_velo = in_sizes[1] / D_DIM;

    // workspace layout (f32 elements)
    float* ws     = (float*)d_ws;
    float* sgm_d  = ws;                       // 16*128*1024 = 2097152
    float* velo_d = sgm_d  + 2097152;         // 16*256*1024 = 4194304
    float* qb     = velo_d + 4194304;         // 2097152
    float* kb     = qb     + 2097152;         // 4194304
    float* v1b    = kb     + 4194304;         // 16*256*512  = 2097152
    float* attnb  = v1b    + 2097152;         // 16*128*256  = 524288
    float* xb     = attnb  + 524288;          // 16*131072   = 2097152
    float* y3a    = xb     + 2097152;         // 16*1024
    float* x3a    = y3a    + 16384;           // 16*1024
    float* x3b    = x3a    + 16384;           // 16*1024
    // bf16 pre-swizzled weights
    __bf16* Wq_s  = (__bf16*)(x3b + 16384);   // 1024*1024
    __bf16* Wk_s  = Wq_s  + 1048576;          // 1024*1024
    __bf16* Wv1_s = Wk_s  + 1048576;          // 1024*512
    __bf16* Wv2_s = Wv1_s + 524288;           // 1024*512
    __bf16* W3b_s = Wv2_s + 524288;           // 1024*1024

    // 0) one-time weight swizzle+convert (f32 -> bf16 B-fragment layout)
    swizzle_w_kernel<<<256, 256, 0, stream>>>(Wq,  Wq_s,  1024, 1024);
    swizzle_w_kernel<<<256, 256, 0, stream>>>(Wk,  Wk_s,  1024, 1024);
    swizzle_w_kernel<<<128, 256, 0, stream>>>(Wv1, Wv1_s, 1024, 512);
    swizzle_w_kernel<<<128, 256, 0, stream>>>(Wv2, Wv2_s, 1024, 512);
    swizzle_w_kernel<<<256, 256, 0, stream>>>(W3b, W3b_s, 1024, 1024);

    // 1) ragged -> dense (gather, zero-padded)
    flatten_kernel<<<B_BATCH * S_LOC, 256, 0, stream>>>(sgm_feats,  sgm_bidx,  n_sgm,  S_LOC, sgm_d);
    flatten_kernel<<<B_BATCH * T_LOC, 256, 0, stream>>>(velo_feats, velo_bidx, n_velo, T_LOC, velo_d);

    // 2) q,k,v1,v2 projections fused with LN+ReLU
    gemm_ln_relu_kernel<1024, 8, 256><<<128, 256, 0, stream>>>(
        sgm_d, D_DIM, Wq_s, gq, bq, qb, 1024, 0, D_DIM);
    gemm_ln_relu_kernel<1024, 8, 256><<<256, 256, 0, stream>>>(
        velo_d, D_DIM, Wk_s, gk, bk, kb, 1024, 0, D_DIM);
    gemm_ln_relu_kernel<512, 4, 256><<<256, 256, 0, stream>>>(
        velo_d, D_DIM, Wv1_s, gv1, bv1, v1b, 512, 0, D_DIM);
    // v2 written straight into x[b, s*1024 + 0..511]
    gemm_ln_relu_kernel<512, 4, 256><<<128, 256, 0, stream>>>(
        sgm_d, D_DIM, Wv2_s, gv2, bv2, xb, 1024, 0, D_DIM);

    // 3) attention scores + softmax, then ctx into x[b, s*1024 + 512..1023]
    attn_kernel<<<dim3(S_LOC / 16, B_BATCH), 256, 0, stream>>>(qb, kb, attnb);
    ctx_kernel<<<dim3(S_LOC / 16, B_BATCH), 256, 0, stream>>>(attnb, v1b, xb);

    // 4) big W3a GEMM: split-K across 256 workgroups (TDM-staged A) + atomics
    zero_kernel<<<64, 256, 0, stream>>>(y3a, 16 * 1024);
    gemm3a_splitk_kernel<512, 256><<<256, 256, 0, stream>>>(xb, W3a, y3a);
    ln_relu_rows_kernel<<<1, 256, 0, stream>>>(y3a, g3a, b3a, x3a, 1024);

    // 5) W3b GEMM fused LN+ReLU (single 16-row tile)
    gemm_ln_relu_kernel<1024, 8, 256><<<1, 256, 0, stream>>>(
        x3a, D_DIM, W3b_s, g3b, b3b, x3b, 1024, 0, D_DIM);

    // 6) heads: quat (normalized) + trans
    heads_kernel<<<B_BATCH, 32, 0, stream>>>(x3b, Wr, br, Wt, bt, (float*)d_out);
}
